// MAMCA_28217935135091
// MI455X (gfx1250) — matrix operand order repaired and checked by reading, not compiled, not run on hardware
//
#include <hip/hip_runtime.h>
#include <hip/hip_bf16.h>

// ---------------------------------------------------------------------------
// Types for CDNA5 WMMA
// ---------------------------------------------------------------------------
typedef __attribute__((ext_vector_type(16))) _Float16 v16h;
typedef __attribute__((ext_vector_type(8)))  _Float16 v8h;
typedef __attribute__((ext_vector_type(8)))  float    v8f;

union Frag16 { v16h v; v8h h[2]; };

#define B_SZ   16
#define L_SEQ  2048
#define M_ROWS (B_SZ * L_SEQ)   // 32768
#define D_MOD  64
#define D_IN   128
#define D_ST   16
#define DT_R   4
#define N_LAY  4

// ---------------------------------------------------------------------------
// Generic WMMA GEMM:  C[m,n] (+)= sum_k A16[m,k] * W[n,k]
//   A16 : M x K  (_Float16, row-major, ld = K)
//   W   : N x K  (float,    row-major, ld = K)  converted to f16 while staging
//   C   : M x ldc (float)
// Block: 256 threads = 8 waves. Block tile 128(M) x 64(N), K staged 64/iter.
// A tile staged with GLOBAL_LOAD_ASYNC_TO_LDS_B128 (ASYNCcnt path), W tile
// converted f32->f16 through VGPRs (overlaps with the async DMA).
// ---------------------------------------------------------------------------
#define BM 128
#define BN 64
#define BK 64
#define BKP (BK + 16)   // +32B pad: keeps 16B alignment, spreads LDS banks

__global__ __launch_bounds__(256) void wmma_gemm_nt(
    const _Float16* __restrict__ A,
    const float*    __restrict__ W,
    float*          __restrict__ C,
    int M, int N, int K, int ldc, int addC)
{
    __shared__ __align__(16) _Float16 As[BM][BKP];
    __shared__ __align__(16) _Float16 Ws[BN][BKP];

    const int tid   = threadIdx.x;
    const int lane  = tid & 31;
    const int wave  = tid >> 5;
    const int mrow  = lane & 15;
    const int ncol  = lane & 15;
    const int khalf = lane >> 4;          // 0: low K half of frag, 1: high

    const int m_blk = blockIdx.x * BM;
    const int n_blk = blockIdx.y * BN;

    v8f acc[4];
#pragma unroll
    for (int t = 0; t < 4; ++t)
#pragma unroll
        for (int j = 0; j < 8; ++j) acc[t][j] = 0.0f;

    // per-thread staging coordinates
    const int ar  = tid >> 1;             // A row within tile
    const int ach = (tid & 1) * 32;       // A col (halfs) within tile
    const int gm  = m_blk + ar;
    const unsigned ldsA = (unsigned)(size_t)&As[ar][ach];   // low 32 bits = LDS offset

    for (int k0 = 0; k0 < K; k0 += BK) {
        // ---- stage A tile: 128 rows x 64 halfs; 2 threads/row, 64B/thread
        if (gm < M) {
            const _Float16* src = A + (size_t)gm * K + k0 + ach;
            unsigned long long ga = (unsigned long long)(size_t)src;
            // async DMA global -> LDS, 16B per lane per op; both LDS and
            // memory addresses take the same INST_OFFSET
            asm volatile(
                "global_load_async_to_lds_b128 %0, %1, off\n\t"
                "global_load_async_to_lds_b128 %0, %1, off offset:16\n\t"
                "global_load_async_to_lds_b128 %0, %1, off offset:32\n\t"
                "global_load_async_to_lds_b128 %0, %1, off offset:48"
                :: "v"(ldsA), "v"(ga) : "memory");
            if (k0 + BK < K) __builtin_prefetch(src + BK, 0, 1);
        } else {
            v8h z = {0,0,0,0,0,0,0,0};
            *(v8h*)&As[ar][ach +  0] = z;
            *(v8h*)&As[ar][ach +  8] = z;
            *(v8h*)&As[ar][ach + 16] = z;
            *(v8h*)&As[ar][ach + 24] = z;
        }
        // ---- stage W tile transposed-in-K: 64 rows(N) x 64 halfs, f32->f16
        {
            const int n  = tid >> 2;
            const int ch = (tid & 3) * 16;
            const int gn = n_blk + n;
            float4 f0, f1, f2, f3;
            if (gn < N) {
                const float4* wp = (const float4*)(W + (size_t)gn * K + k0 + ch);
                f0 = wp[0]; f1 = wp[1]; f2 = wp[2]; f3 = wp[3];
            } else {
                f0 = f1 = f2 = f3 = make_float4(0.f, 0.f, 0.f, 0.f);
            }
            _Float16* d = &Ws[n][ch];
            d[ 0]=(_Float16)f0.x; d[ 1]=(_Float16)f0.y; d[ 2]=(_Float16)f0.z; d[ 3]=(_Float16)f0.w;
            d[ 4]=(_Float16)f1.x; d[ 5]=(_Float16)f1.y; d[ 6]=(_Float16)f1.z; d[ 7]=(_Float16)f1.w;
            d[ 8]=(_Float16)f2.x; d[ 9]=(_Float16)f2.y; d[10]=(_Float16)f2.z; d[11]=(_Float16)f2.w;
            d[12]=(_Float16)f3.x; d[13]=(_Float16)f3.y; d[14]=(_Float16)f3.z; d[15]=(_Float16)f3.w;
        }
        // drain the async DMA before the workgroup barrier (DScnt drained by
        // the barrier lowering itself)
        asm volatile("s_wait_asynccnt 0x0" ::: "memory");
        __syncthreads();

        const int aRow = (wave << 4) + mrow;
#pragma unroll
        for (int kk = 0; kk < BK; kk += 32) {
            // A fragment (16-bit A 16x32 layout: lanes 0-15 K=0..7/16..23,
            // lanes 16-31 K=8..15/24..31, M = lane&15)
            Frag16 af;
            af.h[0] = *(const v8h*)&As[aRow][kk + (khalf << 3)];
            af.h[1] = *(const v8h*)&As[aRow][kk + 16 + (khalf << 3)];
            // load ALL four B fragments into independent registers first so
            // the four WMMAs can issue back-to-back without per-op dscnt waits
            Frag16 bf[4];
#pragma unroll
            for (int t = 0; t < 4; ++t) {
                bf[t].h[0] = *(const v8h*)&Ws[(t << 4) + ncol][kk + (khalf << 3)];
                bf[t].h[1] = *(const v8h*)&Ws[(t << 4) + ncol][kk + 16 + (khalf << 3)];
            }
#pragma unroll
            for (int t = 0; t < 4; ++t) {
                acc[t] = __builtin_amdgcn_wmma_f32_16x16x32_f16(
                    false, af.v, false, bf[t].v, (short)0, acc[t], false, false);
            }
        }
        __syncthreads();
    }

    // ---- store: C layout = 8 VGPRs; lanes 0-15 M=j, lanes 16-31 M=j+8
    const int rowBase = m_blk + (wave << 4) + (khalf << 3);
#pragma unroll
    for (int t = 0; t < 4; ++t) {
        const int col = n_blk + (t << 4) + ncol;
        if (col >= N) continue;
#pragma unroll
        for (int j = 0; j < 8; ++j) {
            const int row = rowBase + j;
            if (row < M) {
                float v = acc[t][j];
                size_t idx = (size_t)row * ldc + col;
                if (addC) v += C[idx];
                C[idx] = v;
            }
        }
    }
}

// ---------------------------------------------------------------------------
// Front-end conv1: (B,1,L) -> (B,64,L), K=3 pad=1, +bias, BN-affine, ReLU
// ---------------------------------------------------------------------------
__global__ void front_conv1(const float* __restrict__ x, const float* __restrict__ w,
                            const float* __restrict__ bias, const float* __restrict__ g,
                            const float* __restrict__ be, float* __restrict__ h1)
{
    int i = blockIdx.x * blockDim.x + threadIdx.x;            // B*64*L
    if (i >= B_SZ * D_MOD * L_SEQ) return;
    int l = i & (L_SEQ - 1), c = (i >> 11) & 63, b = i >> 17;
    float acc = 0.f;
#pragma unroll
    for (int k = 0; k < 3; ++k) {
        int lj = l + k - 1;
        if (lj >= 0 && lj < L_SEQ) acc += w[c * 3 + k] * x[b * L_SEQ + lj];
    }
    float s = g[c] * rsqrtf(1.f + 1e-5f);
    float v = (acc + bias[c]) * s + be[c];
    h1[i] = fmaxf(v, 0.f);
}

// conv2: (B,64,L) -> (B,64,L) then transpose-store into res (B,L,64)
__global__ void front_conv2(const float* __restrict__ h1, const float* __restrict__ w,
                            const float* __restrict__ bias, const float* __restrict__ g,
                            const float* __restrict__ be, float* __restrict__ res)
{
    int i = blockIdx.x * blockDim.x + threadIdx.x;
    if (i >= B_SZ * D_MOD * L_SEQ) return;
    int l = i & (L_SEQ - 1), c = (i >> 11) & 63, b = i >> 17;
    float acc = 0.f;
#pragma unroll
    for (int k = 0; k < 3; ++k) {
        int lj = l + k - 1;
        if (lj < 0 || lj >= L_SEQ) continue;
        const float* ip = h1 + (size_t)b * D_MOD * L_SEQ + lj;
        const float* wp = w + c * D_MOD * 3 + k;
        for (int i2 = 0; i2 < D_MOD; ++i2) acc += wp[i2 * 3] * ip[i2 * L_SEQ];
    }
    float s = g[c] * rsqrtf(1.f + 1e-5f);
    float v = (acc + bias[c]) * s + be[c];
    res[((size_t)((b << 11) + l)) * D_MOD + c] = fmaxf(v, 0.f);
}

// ---------------------------------------------------------------------------
// LayerNorm over width-64 rows: one wave per row (wave32 -> 2 elems/lane).
// Optionally writes f16 row-major and/or f32 transposed (B,64,L).
// ---------------------------------------------------------------------------
__global__ __launch_bounds__(256) void ln_kernel(
    const float* __restrict__ X, const float* __restrict__ g, const float* __restrict__ bb,
    _Float16* __restrict__ out16, float* __restrict__ outT, int Mrows)
{
    int wave = threadIdx.x >> 5;
    int lane = threadIdx.x & 31;
    int row  = blockIdx.x * 8 + wave;
    if (row >= Mrows) return;
    const float* xr = X + (size_t)row * 64;
    float v0 = xr[lane], v1 = xr[lane + 32];
    float s = v0 + v1, q = v0 * v0 + v1 * v1;
#pragma unroll
    for (int o = 16; o >= 1; o >>= 1) {
        s += __shfl_xor(s, o, 32);
        q += __shfl_xor(q, o, 32);
    }
    float mean = s * (1.f / 64.f);
    float var  = q * (1.f / 64.f) - mean * mean;
    float rs   = rsqrtf(var + 1e-5f);
    float y0 = (v0 - mean) * rs * g[lane]      + bb[lane];
    float y1 = (v1 - mean) * rs * g[lane + 32] + bb[lane + 32];
    if (out16) {
        out16[(size_t)row * 64 + lane]      = (_Float16)y0;
        out16[(size_t)row * 64 + lane + 32] = (_Float16)y1;
    }
    if (outT) {
        int b = row >> 11, l = row & (L_SEQ - 1);
        outT[((size_t)(b * 64 + lane))      * L_SEQ + l] = y0;
        outT[((size_t)(b * 64 + lane + 32)) * L_SEQ + l] = y1;
    }
}

// ---------------------------------------------------------------------------
// Depthwise causal conv (K=4, left pad 3) + bias + SiLU; writes f32 + f16
// xz is M x 256, xi channels are cols 0..127
// ---------------------------------------------------------------------------
__global__ void dwconv_silu(const float* __restrict__ xz, const float* __restrict__ cw,
                            const float* __restrict__ cb, float* __restrict__ xi,
                            _Float16* __restrict__ xi16)
{
    int i = blockIdx.x * blockDim.x + threadIdx.x;            // M*128
    if (i >= M_ROWS * D_IN) return;
    int c = i & 127, m = i >> 7;
    int l = m & (L_SEQ - 1);
    float acc = cb[c];
#pragma unroll
    for (int k = 0; k < 4; ++k) {
        int lj = l - 3 + k;
        if (lj >= 0) acc += cw[c * 4 + k] * xz[(size_t)(m - 3 + k) * 256 + c];
    }
    float sv = acc / (1.f + __expf(-acc));
    xi[i] = sv;
    xi16[i] = (_Float16)sv;
}

// dt = softplus(xdbl[:, :4] @ dtw^T + dtb)
__global__ void dt_kernel(const float* __restrict__ xdbl, const float* __restrict__ dtw,
                          const float* __restrict__ dtb, float* __restrict__ dt)
{
    int i = blockIdx.x * blockDim.x + threadIdx.x;
    if (i >= M_ROWS * D_IN) return;
    int d = i & 127, m = i >> 7;
    const float* xr = xdbl + (size_t)m * 36;
    float acc = dtb[d];
#pragma unroll
    for (int r = 0; r < 4; ++r) acc += xr[r] * dtw[d * 4 + r];
    dt[i] = (acc > 20.f) ? acc : log1pf(__expf(acc));
}

// ---------------------------------------------------------------------------
// Selective scan: lane = state index (16), 2 channels per wave32.
// Butterfly shuffle reduces sum_s h*C within each 16-lane group.
// ---------------------------------------------------------------------------
__global__ __launch_bounds__(256) void scan_kernel(
    const float* __restrict__ dtb, const float* __restrict__ xi,
    const float* __restrict__ xdbl, const float* __restrict__ A_log,
    float* __restrict__ y)
{
    int lane = threadIdx.x & 31;
    int gw   = (blockIdx.x * blockDim.x + threadIdx.x) >> 5;  // 0..1023
    int s    = lane & 15;
    int half = lane >> 4;
    int b    = gw >> 6;                                       // 64 waves / batch
    int dch  = ((gw & 63) << 1) | half;                       // channel 0..127
    float a  = -__expf(A_log[dch * 16 + s]);
    float h  = 0.f;
    size_t base = (size_t)b * L_SEQ;
    for (int l = 0; l < L_SEQ; ++l) {
        size_t m  = base + l;
        float dt = dtb[m * 128 + dch];
        float xv = xi [m * 128 + dch];
        float Bv = xdbl[m * 36 + 4  + s];
        float Cv = xdbl[m * 36 + 20 + s];
        h = __expf(dt * a) * h + (dt * xv) * Bv;
        float p = h * Cv;
        p += __shfl_xor(p, 1, 32);
        p += __shfl_xor(p, 2, 32);
        p += __shfl_xor(p, 4, 32);
        p += __shfl_xor(p, 8, 32);
        if (s == 0) y[m * 128 + dch] = p;
    }
}

// gated = (y + xi*Dp) * silu(z); f16 out for out_proj GEMM
__global__ void gating_kernel(const float* __restrict__ y, const float* __restrict__ xi,
                              const float* __restrict__ xz, const float* __restrict__ Dp,
                              _Float16* __restrict__ g16)
{
    int i = blockIdx.x * blockDim.x + threadIdx.x;
    if (i >= M_ROWS * D_IN) return;
    int c = i & 127, m = i >> 7;
    float z  = xz[(size_t)m * 256 + 128 + c];
    float sz = z / (1.f + __expf(-z));
    g16[i] = (_Float16)((y[i] + xi[i] * Dp[c]) * sz);
}

// ---------------------------------------------------------------------------
// Classifier conv: (B,64,Lin) -> (B,64,Lout), K=5 stride=2 pad=2, BN+ReLU
// ---------------------------------------------------------------------------
__global__ void cls_conv(const float* __restrict__ in, const float* __restrict__ w,
                         const float* __restrict__ bias, const float* __restrict__ g,
                         const float* __restrict__ be, float* __restrict__ out,
                         int Lin, int Lout)
{
    int i = blockIdx.x * blockDim.x + threadIdx.x;
    if (i >= B_SZ * 64 * Lout) return;
    int lo = i % Lout;
    int c  = (i / Lout) & 63;
    int b  = i / (Lout * 64);
    float acc = 0.f;
#pragma unroll
    for (int k = 0; k < 5; ++k) {
        int li = 2 * lo + k - 2;
        if (li < 0 || li >= Lin) continue;
        const float* ip = in + (size_t)(b * 64) * Lin + li;
        const float* wp = w + c * 64 * 5 + k;
        for (int i2 = 0; i2 < 64; ++i2) acc += wp[i2 * 5] * ip[(size_t)i2 * Lin];
    }
    float s = g[c] * rsqrtf(1.f + 1e-5f);
    float v = (acc + bias[c]) * s + be[c];
    out[((size_t)(b * 64 + c)) * Lout + lo] = fmaxf(v, 0.f);
}

// pool groups of 4 along L=128 -> (B, 64*32) f16
__global__ void pool_kernel(const float* __restrict__ c4, _Float16* __restrict__ p16)
{
    int i = blockIdx.x * blockDim.x + threadIdx.x;           // B*64*32
    if (i >= B_SZ * 64 * 32) return;
    int gq = i & 31, c = (i >> 5) & 63, b = i >> 11;
    const float* p = c4 + ((size_t)(b * 64 + c)) * 128 + gq * 4;
    p16[(size_t)b * 2048 + c * 32 + gq] = (_Float16)(0.25f * (p[0] + p[1] + p[2] + p[3]));
}

// f = relu((f + fc1_b) * (fc_g/sqrt(1+eps)) + fc_be), in place
__global__ void fc_bnrelu(float* __restrict__ f, const float* __restrict__ b1,
                          const float* __restrict__ g, const float* __restrict__ be)
{
    int i = blockIdx.x * blockDim.x + threadIdx.x;
    if (i >= B_SZ * 256) return;
    int n = i & 255;
    float v = (f[i] + b1[n]) * (g[n] * rsqrtf(1.f + 1e-5f)) + be[n];
    f[i] = fmaxf(v, 0.f);
}

__global__ void fc2_kernel(const float* __restrict__ f, const float* __restrict__ w,
                           const float* __restrict__ b, float* __restrict__ out)
{
    int i = blockIdx.x * blockDim.x + threadIdx.x;           // B*5
    if (i >= B_SZ * 5) return;
    int o = i % 5, bi = i / 5;
    float acc = b[o];
    for (int k = 0; k < 256; ++k) acc += f[bi * 256 + k] * w[o * 256 + k];
    out[i] = acc;
}

// ---------------------------------------------------------------------------
// Launch
// ---------------------------------------------------------------------------
static inline size_t al256(size_t x) { return (x + 255) & ~(size_t)255; }

extern "C" void kernel_launch(void* const* d_in, const int* in_sizes, int n_in,
                              void* d_out, int out_size, void* d_ws, size_t ws_size,
                              hipStream_t stream)
{
    const float* x        = (const float*)d_in[0];
    const float* den_w1   = (const float*)d_in[1];
    const float* den_b1   = (const float*)d_in[2];
    const float* den_g1   = (const float*)d_in[3];
    const float* den_be1  = (const float*)d_in[4];
    const float* den_w2   = (const float*)d_in[5];
    const float* den_b2   = (const float*)d_in[6];
    const float* den_g2   = (const float*)d_in[7];
    const float* den_be2  = (const float*)d_in[8];
    const float* ln_g     = (const float*)d_in[9];
    const float* ln_b     = (const float*)d_in[10];
    const float* in_w     = (const float*)d_in[11];
    const float* conv_w   = (const float*)d_in[12];
    const float* conv_b   = (const float*)d_in[13];
    const float* xproj_w  = (const float*)d_in[14];
    const float* dtproj_w = (const float*)d_in[15];
    const float* dtproj_b = (const float*)d_in[16];
    const float* A_log    = (const float*)d_in[17];
    const float* Dp       = (const float*)d_in[18];
    const float* out_w    = (const float*)d_in[19];
    const float* fln_g    = (const float*)d_in[20];
    const float* fln_b    = (const float*)d_in[21];
    const float* cls_w1   = (const float*)d_in[22];
    const float* cls_b1   = (const float*)d_in[23];
    const float* cls_g1   = (const float*)d_in[24];
    const float* cls_be1  = (const float*)d_in[25];
    const float* cls_wR   = (const float*)d_in[26];
    const float* cls_bR   = (const float*)d_in[27];
    const float* cls_gR   = (const float*)d_in[28];
    const float* cls_beR  = (const float*)d_in[29];
    const float* fc1_w    = (const float*)d_in[30];
    const float* fc1_b    = (const float*)d_in[31];
    const float* fc_g     = (const float*)d_in[32];
    const float* fc_be    = (const float*)d_in[33];
    const float* fc2_w    = (const float*)d_in[34];
    const float* fc2_b    = (const float*)d_in[35];

    char* ws = (char*)d_ws;
    size_t off = 0;
    auto wsa = [&](size_t bytes) { void* p = ws + off; off = al256(off + bytes); return p; };

    float*     res    = (float*)    wsa((size_t)M_ROWS * 64 * 4);
    float*     h1     = (float*)    wsa((size_t)B_SZ * 64 * L_SEQ * 4);
    _Float16*  xln16  = (_Float16*) wsa((size_t)M_ROWS * 64 * 2);
    float*     xz     = (float*)    wsa((size_t)M_ROWS * 256 * 4);
    float*     xibuf  = (float*)    wsa((size_t)M_ROWS * 128 * 4);
    _Float16*  xi16   = (_Float16*) wsa((size_t)M_ROWS * 128 * 2);
    float*     xdbl   = (float*)    wsa((size_t)M_ROWS * 36 * 4);
    float*     dtbuf  = (float*)    wsa((size_t)M_ROWS * 128 * 4);
    float*     ybuf   = (float*)    wsa((size_t)M_ROWS * 128 * 4);
    _Float16*  g16    = (_Float16*) wsa((size_t)M_ROWS * 128 * 2);
    float*     hT     = (float*)    wsa((size_t)B_SZ * 64 * L_SEQ * 4);
    float*     c1     = (float*)    wsa((size_t)B_SZ * 64 * 1024 * 4);
    float*     c2     = (float*)    wsa((size_t)B_SZ * 64 * 512 * 4);
    float*     c3     = (float*)    wsa((size_t)B_SZ * 64 * 256 * 4);
    float*     c4     = (float*)    wsa((size_t)B_SZ * 64 * 128 * 4);
    _Float16*  pool16 = (_Float16*) wsa((size_t)B_SZ * 2048 * 2);
    float*     fcraw  = (float*)    wsa((size_t)B_SZ * 256 * 4);

    const dim3 blk(256);
    const int elemBC = B_SZ * 64 * L_SEQ;            // 2M
    const int elemDI = M_ROWS * D_IN;                // 4M

    // Front-end
    front_conv1<<<(elemBC + 255) / 256, blk, 0, stream>>>(x, den_w1, den_b1, den_g1, den_be1, h1);
    front_conv2<<<(elemBC + 255) / 256, blk, 0, stream>>>(h1, den_w2, den_b2, den_g2, den_be2, res);

    // Mamba layers
    for (int i = 0; i < N_LAY; ++i) {
        const float* lg   = ln_g     + i * 64;
        const float* lb   = ln_b     + i * 64;
        const float* iw   = in_w     + (size_t)i * 256 * 64;
        const float* cw   = conv_w   + (size_t)i * 128 * 4;
        const float* cb   = conv_b   + (size_t)i * 128;
        const float* xw   = xproj_w  + (size_t)i * 36 * 128;
        const float* dtw  = dtproj_w + (size_t)i * 128 * 4;
        const float* dtbv = dtproj_b + (size_t)i * 128;
        const float* al   = A_log    + (size_t)i * 128 * 16;
        const float* dp   = Dp       + (size_t)i * 128;
        const float* ow   = out_w    + (size_t)i * 64 * 128;

        ln_kernel<<<M_ROWS / 8, blk, 0, stream>>>(res, lg, lb, xln16, (float*)nullptr, M_ROWS);

        wmma_gemm_nt<<<dim3(M_ROWS / BM, 256 / BN), blk, 0, stream>>>(
            xln16, iw, xz, M_ROWS, 256, 64, 256, 0);

        dwconv_silu<<<(elemDI + 255) / 256, blk, 0, stream>>>(xz, cw, cb, xibuf, xi16);

        wmma_gemm_nt<<<dim3(M_ROWS / BM, 1), blk, 0, stream>>>(
            xi16, xw, xdbl, M_ROWS, 36, 128, 36, 0);

        dt_kernel<<<(elemDI + 255) / 256, blk, 0, stream>>>(xdbl, dtw, dtbv, dtbuf);

        scan_kernel<<<(B_SZ * 64 * 32) / 256, blk, 0, stream>>>(dtbuf, xibuf, xdbl, al, ybuf);

        gating_kernel<<<(elemDI + 255) / 256, blk, 0, stream>>>(ybuf, xibuf, xz, dp, g16);

        // residual: res += gated @ out_w^T
        wmma_gemm_nt<<<dim3(M_ROWS / BM, 1), blk, 0, stream>>>(
            g16, ow, res, M_ROWS, 64, 128, 64, 1);
    }

    // Final LN, transposed into (B,64,L)
    ln_kernel<<<M_ROWS / 8, blk, 0, stream>>>(res, fln_g, fln_b, (_Float16*)nullptr, hT, M_ROWS);

    // Classifier convs (stride 2, L: 2048->1024->512->256->128)
    cls_conv<<<(B_SZ * 64 * 1024 + 255) / 256, blk, 0, stream>>>(hT, cls_w1, cls_b1, cls_g1, cls_be1, c1, 2048, 1024);
    cls_conv<<<(B_SZ * 64 * 512  + 255) / 256, blk, 0, stream>>>(c1, cls_wR + 0 * 64 * 64 * 5, cls_bR + 0 * 64, cls_gR + 0 * 64, cls_beR + 0 * 64, c2, 1024, 512);
    cls_conv<<<(B_SZ * 64 * 256  + 255) / 256, blk, 0, stream>>>(c2, cls_wR + 1 * 64 * 64 * 5, cls_bR + 1 * 64, cls_gR + 1 * 64, cls_beR + 1 * 64, c3, 512, 256);
    cls_conv<<<(B_SZ * 64 * 128  + 255) / 256, blk, 0, stream>>>(c3, cls_wR + 2 * 64 * 64 * 5, cls_bR + 2 * 64, cls_gR + 2 * 64, cls_beR + 2 * 64, c4, 256, 128);

    pool_kernel<<<(B_SZ * 64 * 32 + 255) / 256, blk, 0, stream>>>(c4, pool16);

    // fc1 GEMM (M=16, N=256, K=2048) on WMMA path
    wmma_gemm_nt<<<dim3(1, 256 / BN), blk, 0, stream>>>(pool16, fc1_w, fcraw, B_SZ, 256, 2048, 256, 0);

    fc_bnrelu<<<(B_SZ * 256 + 255) / 256, blk, 0, stream>>>(fcraw, fc1_b, fc_g, fc_be);
    fc2_kernel<<<1, 128, 0, stream>>>(fcraw, fc2_w, fc2_b, (float*)d_out);
}
